// PrototypeField_13357348290854
// MI455X (gfx1250) — matrix operand order repaired and checked by reading, not compiled, not run on hardware
//
#include <hip/hip_runtime.h>
#include <hip/hip_bf16.h>

// PrototypeField fused: segment-sum via one-hot x dirs_n WMMA (bf16 in, f32 acc),
// loss = 0.1*(1 - (1/B) * sum_c dot(new_v_c, sums_c))  [exact algebraic identity]

#define CLS   128
#define DIM   256
#define SLAB  32
#define WAVES 16
#define TPB   (WAVES * 32)

typedef __attribute__((ext_vector_type(16))) __bf16 v16bf;
typedef __attribute__((ext_vector_type(8)))  float  v8f;

struct BPack { uint4 lo, hi; };

__device__ __forceinline__ float wave_sum(float x) {
#pragma unroll
  for (int o = 16; o > 0; o >>= 1) x += __shfl_xor(x, o, 32);
  return x;
}

__device__ __forceinline__ unsigned short bf16_bits(float f) {
  return __builtin_bit_cast(unsigned short, (__bf16)f);
}

// LDS B-operand layout: [ping][g][lt][kk] in u16 units, padded strides so that
// producer u16 scatter-stores and consumer ds_load_b128 are <=2-way conflicted.
#define LT_STRIDE 24                      // 16 used + 8 pad u16 per (g,lane) slot
#define G_STRIDE  (32 * LT_STRIDE + 8)    // 776 u16 per 16-dim group
#define P_STRIDE  (16 * G_STRIDE)         // per ping buffer

__device__ __forceinline__ int bidx(int ping, int g, int lt, int kk) {
  return ping * P_STRIDE + g * G_STRIDE + lt * LT_STRIDE + kk;
}

// ---------------------------------------------------------------------------
// Kernel 1: stream dirs, normalize rows, WMMA one-hot segment-sum into
// per-workgroup partial sums (disjoint writes, fully deterministic).
// Wave w: M-tile = w>>1 (16 classes), N-half = w&1 (128 dims = 8 WMMA N-tiles).
// ---------------------------------------------------------------------------
__global__ __launch_bounds__(TPB) void seg_sum_wmma(
    const float* __restrict__ dirs, const int* __restrict__ labels,
    float* __restrict__ part, int* __restrict__ cnt, int slabs)
{
  __shared__ unsigned short ldsB[2 * P_STRIDE];
  __shared__ int ldsLab[2][SLAB];
  __shared__ int hist[CLS];

  const int tid   = threadIdx.x;
  const int wave  = tid >> 5;
  const int lane  = tid & 31;
  const int wg    = blockIdx.x;
  const int mtile = wave >> 1;
  const int nhalf = wave & 1;
  const int mbase = mtile * 16;
  const long rowBase = (long)wg * slabs * SLAB;
  const int nrows = slabs * SLAB;

  // ---- deterministic per-WG label histogram (integer LDS atomics) ----
  for (int i = tid; i < CLS; i += TPB) hist[i] = 0;
  __syncthreads();
  for (int i = tid; i < nrows; i += TPB)
    atomicAdd(&hist[labels[rowBase + i]], 1);
  __syncthreads();
  for (int i = tid; i < CLS; i += TPB) cnt[wg * CLS + i] = hist[i];

  v8f acc[8] = {};

  const int mrow  = mbase + (lane & 15);
  const int kbase = (lane < 16) ? 0 : 8;

  for (int s = 0; s < slabs; ++s) {
    const int ping = s & 1;

    // ---- producer: this wave normalizes slab rows 2*wave, 2*wave+1 ----
    const int rA = 2 * wave, rB = rA + 1;
    const long gr = rowBase + (long)s * SLAB + rA;
    const float4* pA = (const float4*)(dirs + gr * DIM) + lane * 2;
    const float4* pB = (const float4*)(dirs + (gr + 1) * DIM) + lane * 2;
    float4 a0 = pA[0], a1 = pA[1];
    float4 b0 = pB[0], b1 = pB[1];
    if (s + 1 < slabs)
      __builtin_prefetch(dirs + (gr + SLAB) * DIM + lane * 8, 0, 0);

    float sA = a0.x*a0.x + a0.y*a0.y + a0.z*a0.z + a0.w*a0.w
             + a1.x*a1.x + a1.y*a1.y + a1.z*a1.z + a1.w*a1.w;
    float sB = b0.x*b0.x + b0.y*b0.y + b0.z*b0.z + b0.w*b0.w
             + b1.x*b1.x + b1.y*b1.y + b1.z*b1.z + b1.w*b1.w;
    sA = wave_sum(sA);
    sB = wave_sum(sB);
    const float invA = 1.0f / fmaxf(sqrtf(sA), 1e-12f);
    const float invB = 1.0f / fmaxf(sqrtf(sB), 1e-12f);

    if (tid < SLAB) ldsLab[ping][tid] = labels[rowBase + (long)s * SLAB + tid];

    // lane owns dims d = 8*lane .. 8*lane+7 -> group g = lane>>1, d%16 base:
    const int g  = lane >> 1;
    const int dl = (lane & 1) * 8;
    const int ltA = (rA & 16) + dl;
    const int ltB = (rB & 16) + dl;
    const int kkA = rA & 15, kkB = rB & 15;
    float va[8] = {a0.x, a0.y, a0.z, a0.w, a1.x, a1.y, a1.z, a1.w};
    float vb[8] = {b0.x, b0.y, b0.z, b0.w, b1.x, b1.y, b1.z, b1.w};
#pragma unroll
    for (int j = 0; j < 8; ++j) {
      ldsB[bidx(ping, g, ltA + j, kkA)] = bf16_bits(va[j] * invA);
      ldsB[bidx(ping, g, ltB + j, kkB)] = bf16_bits(vb[j] * invB);
    }

    __syncthreads();   // one barrier/slab; ping-pong covers WAR to s-1 buffer

    // ---- consumer: A = one-hot(labels) tile (registers), B from LDS ----
    v16bf amat;
#pragma unroll
    for (int e = 0; e < 16; ++e) {
      const int k = e + (kbase << 1);  // A layout K mapping (contiguous: k = 16h + e)
      amat[e] = (ldsLab[ping][k] == mrow) ? (__bf16)1.0f : (__bf16)0.0f;
    }
#pragma unroll
    for (int t = 0; t < 8; ++t) {
      const int gg = nhalf * 8 + t;
      const uint4* bp = (const uint4*)&ldsB[bidx(ping, gg, lane, 0)];
      BPack pk; pk.lo = bp[0]; pk.hi = bp[1];
      v16bf bmat = __builtin_bit_cast(v16bf, pk);
      acc[t] = __builtin_amdgcn_wmma_f32_16x16x32_bf16(
                 false, amat, false, bmat, (short)0, acc[t], false, false);
    }
  }

  // ---- per-WG partial sums: disjoint (c,d) region per wave, plain stores ----
  float* base = part + (size_t)wg * CLS * DIM;
  const int crow0 = mbase + ((lane < 16) ? 0 : 8);
#pragma unroll
  for (int t = 0; t < 8; ++t) {
    const int d = nhalf * 128 + t * 16 + (lane & 15);
#pragma unroll
    for (int v = 0; v < 8; ++v)
      base[(size_t)(crow0 + v) * DIM + d] = acc[t][v];
  }
}

// ---------------------------------------------------------------------------
// Kernel 2: deterministic reduction of per-WG partials -> total sums, counts.
// ---------------------------------------------------------------------------
__global__ __launch_bounds__(256) void reduce_partials(
    const float* __restrict__ part, const int* __restrict__ cnt,
    float* __restrict__ tot, int* __restrict__ totc, int nwg)
{
  const int idx = blockIdx.x * 256 + threadIdx.x;     // 0 .. CLS*DIM-1
  float s = 0.f;
  for (int w = 0; w < nwg; ++w) s += part[(size_t)w * (CLS * DIM) + idx];
  tot[idx] = s;
  if (blockIdx.x == 0 && threadIdx.x < CLS) {
    int cs = 0;
    for (int w = 0; w < nwg; ++w) cs += cnt[w * CLS + threadIdx.x];
    totc[threadIdx.x] = cs;
  }
}

// ---------------------------------------------------------------------------
// Kernel 3: EMA + double-normalize per class, write new_v, loss from
// sum_c dot(new_v_c, sums_c). One WG, 32 waves, wave handles 4 classes.
// ---------------------------------------------------------------------------
__global__ __launch_bounds__(1024) void finish(
    const float* __restrict__ tot, const int* __restrict__ totc,
    const float* __restrict__ v_class, const int* __restrict__ stepp,
    float* __restrict__ out, int Bn)
{
  __shared__ float dparts[32];
  const int tid = threadIdx.x, wave = tid >> 5, lane = tid & 31;
  const float stepf = (float)stepp[0];
  const float alpha = fminf(0.02f, stepf / (stepf + 50.0f));
  float dacc = 0.0f;

#pragma unroll
  for (int i = 0; i < 4; ++i) {
    const int c = wave * 4 + i;
    const float4* sp = (const float4*)(tot + (size_t)c * DIM) + lane * 2;
    const float4* vp = (const float4*)(v_class + (size_t)c * DIM) + lane * 2;
    float4 s0 = sp[0], s1 = sp[1];
    float4 v0 = vp[0], v1 = vp[1];
    float sv[8] = {s0.x, s0.y, s0.z, s0.w, s1.x, s1.y, s1.z, s1.w};
    float vv[8] = {v0.x, v0.y, v0.z, v0.w, v1.x, v1.y, v1.z, v1.w};
    const int n = totc[c];
    const float invn = 1.0f / fmaxf((float)n, 1.0f);
    float bm[8], ssq = 0.f;
#pragma unroll
    for (int j = 0; j < 8; ++j) { bm[j] = sv[j] * invn; ssq += bm[j] * bm[j]; }
    ssq = wave_sum(ssq);
    const float inv1 = 1.0f / fmaxf(sqrtf(ssq), 1e-12f);
    float u[8], ssq2 = 0.f;
#pragma unroll
    for (int j = 0; j < 8; ++j) {
      u[j] = (1.0f - alpha) * vv[j] + alpha * (bm[j] * inv1);
      ssq2 += u[j] * u[j];
    }
    ssq2 = wave_sum(ssq2);
    const float inv2 = 1.0f / fmaxf(sqrtf(ssq2), 1e-12f);
    float nv[8];
#pragma unroll
    for (int j = 0; j < 8; ++j) {
      nv[j] = (n > 0) ? u[j] * inv2 : vv[j];
      dacc += nv[j] * sv[j];
    }
    float* op = out + 1 + (size_t)c * DIM + lane * 8;
    ((float4*)op)[0] = make_float4(nv[0], nv[1], nv[2], nv[3]);
    ((float4*)op)[1] = make_float4(nv[4], nv[5], nv[6], nv[7]);
  }
  dacc = wave_sum(dacc);
  if (lane == 0) dparts[wave] = dacc;
  __syncthreads();
  if (wave == 0) {
    float x = wave_sum(dparts[lane]);
    if (lane == 0) out[0] = 0.1f * (1.0f - x / (float)Bn);
  }
}

// ---------------------------------------------------------------------------
extern "C" void kernel_launch(void* const* d_in, const int* in_sizes, int n_in,
                              void* d_out, int out_size, void* d_ws, size_t ws_size,
                              hipStream_t stream) {
  const float* dirs    = (const float*)d_in[0];
  const float* v_class = (const float*)d_in[1];
  const int*   labels  = (const int*)d_in[2];
  const int*   stepp   = (const int*)d_in[3];
  float* out = (float*)d_out;
  const int Bn = in_sizes[0] / DIM;   // 262144

  // choose #workgroups so that partial buffers fit the workspace
  int nwg = 64;
  while (nwg > 1) {
    const size_t need = (size_t)nwg * CLS * DIM * 4    // partial sums
                      + (size_t)nwg * CLS * 4          // partial counts
                      + (size_t)CLS * DIM * 4          // total sums
                      + (size_t)CLS * 4;               // total counts
    if (need <= ws_size && (Bn % (nwg * SLAB)) == 0) break;
    nwg >>= 1;
  }

  char* w = (char*)d_ws;
  float* part = (float*)w;
  int*   cnt  = (int*)(w + (size_t)nwg * CLS * DIM * 4);
  float* tot  = (float*)((char*)cnt + (size_t)nwg * CLS * 4);
  int*   totc = (int*)((char*)tot + (size_t)CLS * DIM * 4);

  const int slabs = Bn / (nwg * SLAB);
  seg_sum_wmma<<<nwg, TPB, 0, stream>>>(dirs, labels, part, cnt, slabs);
  reduce_partials<<<(CLS * DIM) / 256, 256, 0, stream>>>(part, cnt, tot, totc, nwg);
  finish<<<1, 1024, 0, stream>>>(tot, totc, v_class, stepp, out, Bn);
}
